// GGVQVAE_91079076479687
// MI455X (gfx1250) — matrix operand order repaired and checked by reading, not compiled, not run on hardware
//
#include <hip/hip_runtime.h>
#include <hip/hip_bf16.h>
#include <math.h>

// ---------------------------------------------------------------------------
// CDNA5 (gfx1250) VQ-VAE forward.
//  - Weights pre-packed (bf16, WMMA A-fragment order) once per launch:
//    each lane loads its A fragment as one coalesced 32B global load.
//  - Convs: LDS-staged implicit GEMM, block tile 128(M) x 64(N), K-step 32,
//    double-buffered B tile (1 barrier / K-step, load/WMMA overlap).
//  - VQ argmin: packed codebook staged to LDS via TDM (tensor_load_to_lds),
//    then 2 WMMAs per 16-code tile straight from LDS fragments.
// ---------------------------------------------------------------------------

typedef __attribute__((ext_vector_type(16))) __bf16          v16bf;
typedef __attribute__((ext_vector_type(16))) unsigned short  v16u;
typedef __attribute__((ext_vector_type(8)))  float           v8f;
typedef __attribute__((ext_vector_type(4)))  unsigned int    u32x4;
typedef __attribute__((ext_vector_type(8)))  int             i32x8;
typedef __attribute__((ext_vector_type(4)))  int             i32x4;

__device__ __forceinline__ unsigned short f2bf(float f) {
    unsigned int u = __builtin_bit_cast(unsigned int, f);
    unsigned int r = u + 0x7fffu + ((u >> 16) & 1u);   // round-to-nearest-even
    return (unsigned short)(r >> 16);
}
__device__ __forceinline__ float lrelu(float v) { return v > 0.f ? v : 0.01f * v; }

// A-fragment K index (ISA 7.12.2): k = h + (h&8) + 8*laneHi
__device__ __forceinline__ int a_k(int h, int hi8) { return h + (h & 8) + hi8; }

// ---------------------------------------------------------------------------
// Weight packing: P[mt][ks][lane][h] = bf16(W[m = mt*16+lane%16][k = ks*32+a_k]).
// Zero-filled outside [Cout,KK] so fragment loads are always valid.
// ---------------------------------------------------------------------------
__global__ void pack_conv_w(const float* __restrict__ W, unsigned short* __restrict__ P,
                            int Cout, int KK, int mtPad, int kSteps)
{
    int idx = blockIdx.x * 256 + threadIdx.x;
    if (idx >= mtPad * kSteps * 512) return;
    int h    = idx & 15;
    int lane = (idx >> 4) & 31;
    int rest = idx >> 9;
    int ks   = rest % kSteps;
    int mt   = rest / kSteps;
    int m = mt * 16 + (lane & 15);
    int k = ks * 32 + a_k(h, (lane >> 4) * 8);
    float v = (m < Cout && k < KK) ? W[(size_t)m * KK + k] : 0.f;
    P[idx] = f2bf(v);
}

// Transpose-conv weights (Cin, Cout, 4, 4), packed per parity class pc=(py,px):
// P[pc][mt][ks][lane][h]; k -> (ci, a, b): kh = (1-py)+2a, kw = (1-px)+2b.
__global__ void pack_tconv_w(const float* __restrict__ W, unsigned short* __restrict__ P,
                             int Cin, int Cout, int mtPad, int kSteps)
{
    int idx = blockIdx.x * 256 + threadIdx.x;
    if (idx >= 4 * mtPad * kSteps * 512) return;
    int h    = idx & 15;
    int lane = (idx >> 4) & 31;
    int rest = idx >> 9;
    int ks   = rest % kSteps;  rest /= kSteps;
    int mt   = rest % mtPad;
    int pc   = rest / mtPad;
    int py = pc >> 1, px = pc & 1;
    int m = mt * 16 + (lane & 15);
    int k = ks * 32 + a_k(h, (lane >> 4) * 8);
    float v = 0.f;
    if (m < Cout && k < Cin * 4) {
        int ci = k >> 2, a = (k >> 1) & 1, bb = k & 1;
        v = W[(((size_t)ci * Cout + m) * 4 + ((1 - py) + 2 * a)) * 4 + ((1 - px) + 2 * bb)];
    }
    P[idx] = f2bf(v);
}

// Codebook packed into B-fragment order: P[ct][kb2][lane][h] = CB[ct*16+lane%16][kb2*32+a_k(h,8*(lane/16))]
__global__ void pack_cb(const float* __restrict__ CB, unsigned short* __restrict__ P,
                        int Kcb, int D)
{
    int idx = blockIdx.x * 256 + threadIdx.x;
    if (idx >= (Kcb / 16) * 2 * 512) return;
    int h    = idx & 15;
    int lane = (idx >> 4) & 31;
    int rest = idx >> 9;
    int kb2  = rest & 1;
    int ct   = rest >> 1;
    float v = CB[(size_t)(ct * 16 + (lane & 15)) * D + kb2 * 32 + a_k(h, (lane >> 4) * 8)];
    P[idx] = f2bf(v);
}

// ---------------------------------------------------------------------------
// B-tile loader: im2col 32 k-rows x 64 pixels -> LDS bf16, two 16-bit stores per pixel pair.
// ---------------------------------------------------------------------------
__device__ __forceinline__ void conv_loadB(
    unsigned short (*sB)[32][16], const float* __restrict__ X,
    int tid, int kb, int KK, int KHW, int KW, int nbase, int Ntot,
    int Cin, int IH, int IW, int OW, int OHOW, int stride, int pad, int preact)
{
#pragma unroll
    for (int i = 0; i < 4; ++i) {
        int pr = tid + i * 256;            // pair index 0..1023
        int kc = pr >> 5;                  // 0..31
        int pn = (pr & 31) * 2;            // 0,2,..,62
        int k = kb + kc;
        unsigned int packed = 0;
        if (k < KK) {
            int ci = k / KHW; int r = k - ci * KHW;
            int kh = r / KW;  int kw = r - kh * KW;
#pragma unroll
            for (int q = 0; q < 2; ++q) {
                int p = nbase + pn + q;
                float xv = 0.f;
                if (p < Ntot) {
                    int b  = p / OHOW; int r2 = p - b * OHOW;
                    int oh = r2 / OW;  int ow = r2 - oh * OW;
                    int ih = oh * stride - pad + kh;
                    int iw = ow * stride - pad + kw;
                    if (ih >= 0 && ih < IH && iw >= 0 && iw < IW) {
                        xv = X[(((size_t)b * Cin + ci) * IH + ih) * IW + iw];
                        if (preact) xv = lrelu(xv);
                    }
                }
                packed |= (unsigned int)f2bf(xv) << (16 * q);
            }
        }
        sB[pn >> 4][(pn & 15) + ((kc >> 3) & 1) * 16][(kc & 7) + ((kc >> 4) << 3)] = (unsigned short)packed;
        sB[pn >> 4][(pn & 15) + 1 + ((kc >> 3) & 1) * 16][(kc & 7) + ((kc >> 4) << 3)] = (unsigned short)(packed >> 16);
    }
}

__device__ __forceinline__ void tconv_loadB(
    unsigned short (*sB)[32][16], const float* __restrict__ X,
    int tid, int kb, int KK, int nbase, int Ntc,
    int Cin, int IH, int IW, int IHW, int py, int px, int preact)
{
#pragma unroll
    for (int i = 0; i < 4; ++i) {
        int pr = tid + i * 256;
        int kc = pr >> 5;
        int pn = (pr & 31) * 2;
        int k = kb + kc;
        unsigned int packed = 0;
        if (k < KK) {
            int ci = k >> 2, a = (k >> 1) & 1, bb = k & 1;
#pragma unroll
            for (int q = 0; q < 2; ++q) {
                int p = nbase + pn + q;
                float xv = 0.f;
                if (p < Ntc) {
                    int b  = p / IHW; int r2 = p - b * IHW;
                    int i2 = r2 / IW; int j = r2 - i2 * IW;
                    int ih = i2 + py - a, iw = j + px - bb;
                    if (ih >= 0 && ih < IH && iw >= 0 && iw < IW) {
                        xv = X[(((size_t)b * Cin + ci) * IH + ih) * IW + iw];
                        if (preact) xv = lrelu(xv);
                    }
                }
                packed |= (unsigned int)f2bf(xv) << (16 * q);
            }
        }
        sB[pn >> 4][(pn & 15) + ((kc >> 3) & 1) * 16][(kc & 7) + ((kc >> 4) << 3)] = (unsigned short)packed;
        sB[pn >> 4][(pn & 15) + 1 + ((kc >> 3) & 1) * 16][(kc & 7) + ((kc >> 4) << 3)] = (unsigned short)(packed >> 16);
    }
}

// ---------------------------------------------------------------------------
// Conv as implicit GEMM. Block: M=128 (8 waves x 16), N=64 (4 WMMA tiles).
// ---------------------------------------------------------------------------
__global__ __launch_bounds__(256) void conv_wmma(
    const float* __restrict__ X, const unsigned short* __restrict__ Wp,
    const float* __restrict__ bias, const float* __restrict__ res,
    float* __restrict__ Y,
    int B, int Cin, int IH, int IW, int Cout, int OH, int OW,
    int KH, int KW, int stride, int pad, int preact, int postact)
{
    __shared__ __align__(32) unsigned short sB[2][4][32][16];

    const int tid  = threadIdx.x;
    const int lane = tid & 31;
    const int wave = tid >> 5;
    const int OHOW = OH * OW;
    const int Ntot = B * OHOW;
    const int ntB  = Ntot >> 6;
    const int mb   = blockIdx.x / ntB;
    const int nb   = blockIdx.x - mb * ntB;
    const int nbase = nb * 64;
    const int KHW = KH * KW;
    const int KK  = Cin * KHW;
    const int kSteps = (KK + 31) >> 5;
    const int laneHi = lane >> 4, lm = lane & 15, hi8 = laneHi * 8;
    const size_t aBase = (size_t)(mb * 8 + wave) * kSteps * 512 + (size_t)lane * 16;

    const v8f vzero = {0.f,0.f,0.f,0.f,0.f,0.f,0.f,0.f};
    v8f acc[4];
#pragma unroll
    for (int t = 0; t < 4; ++t) acc[t] = vzero;

    conv_loadB(sB[0], X, tid, 0, KK, KHW, KW, nbase, Ntot,
               Cin, IH, IW, OW, OHOW, stride, pad, preact);
    for (int ks = 0; ks < kSteps; ++ks) {
        __syncthreads();
        int cur = ks & 1;
        if (ks + 1 < kSteps) {
            __builtin_prefetch(&Wp[aBase + (size_t)(ks + 1) * 512]);
            conv_loadB(sB[cur ^ 1], X, tid, (ks + 1) * 32, KK, KHW, KW, nbase, Ntot,
                       Cin, IH, IW, OW, OHOW, stride, pad, preact);
        }
        v16bf av = __builtin_bit_cast(v16bf, *(const v16u*)&Wp[aBase + (size_t)ks * 512]);
#pragma unroll
        for (int t = 0; t < 4; ++t) {
            v16bf bv = __builtin_bit_cast(v16bf, *(const v16u*)&sB[cur][t][lane][0]);
            acc[t] = __builtin_amdgcn_wmma_f32_16x16x32_bf16(
                false, av, false, bv, (short)0, acc[t], false, false);
        }
    }
#pragma unroll
    for (int r = 0; r < 8; ++r) {
        int m = mb * 128 + wave * 16 + hi8 + r;
        if (m >= Cout) continue;
        float bvl = bias ? bias[m] : 0.f;
#pragma unroll
        for (int t = 0; t < 4; ++t) {
            int p = nbase + t * 16 + lm;
            if (p >= Ntot) continue;
            int b = p / OHOW; int hw = p - b * OHOW;
            size_t off = ((size_t)b * Cout + m) * OHOW + hw;
            float v = acc[t][r] + bvl;
            if (res) v += res[off];
            if (postact == 1) v = lrelu(v);
            else if (postact == 2) v = v > 0.f ? v : 0.f;
            Y[off] = v;
        }
    }
}

// ---------------------------------------------------------------------------
// ConvTranspose2d(k=4,s=2,p=1): 4 parity-class 2x2-tap convs (blockIdx.y).
// postact: 1 lrelu, 3 tanh.
// ---------------------------------------------------------------------------
__global__ __launch_bounds__(256) void tconv_wmma(
    const float* __restrict__ X, const unsigned short* __restrict__ Wp,
    const float* __restrict__ bias, float* __restrict__ Y,
    int B, int Cin, int IH, int IW, int Cout, int mtPad, int preact, int postact)
{
    __shared__ __align__(32) unsigned short sB[2][4][32][16];

    const int tid  = threadIdx.x;
    const int lane = tid & 31;
    const int wave = tid >> 5;
    const int pc = blockIdx.y, py = pc >> 1, px = pc & 1;
    const int IHW = IH * IW;
    const int Ntc = B * IHW;
    const int ntB = Ntc >> 6;
    const int mb  = blockIdx.x / ntB;
    const int nb  = blockIdx.x - mb * ntB;
    const int nbase = nb * 64;
    const int KK = Cin * 4;
    const int kSteps = (KK + 31) >> 5;
    const int laneHi = lane >> 4, lm = lane & 15, hi8 = laneHi * 8;
    const size_t aBase = ((size_t)pc * mtPad + (mb * 8 + wave)) * kSteps * 512
                       + (size_t)lane * 16;

    const v8f vzero = {0.f,0.f,0.f,0.f,0.f,0.f,0.f,0.f};
    v8f acc[4];
#pragma unroll
    for (int t = 0; t < 4; ++t) acc[t] = vzero;

    tconv_loadB(sB[0], X, tid, 0, KK, nbase, Ntc, Cin, IH, IW, IHW, py, px, preact);
    for (int ks = 0; ks < kSteps; ++ks) {
        __syncthreads();
        int cur = ks & 1;
        if (ks + 1 < kSteps) {
            __builtin_prefetch(&Wp[aBase + (size_t)(ks + 1) * 512]);
            tconv_loadB(sB[cur ^ 1], X, tid, (ks + 1) * 32, KK, nbase, Ntc,
                        Cin, IH, IW, IHW, py, px, preact);
        }
        v16bf av = __builtin_bit_cast(v16bf, *(const v16u*)&Wp[aBase + (size_t)ks * 512]);
#pragma unroll
        for (int t = 0; t < 4; ++t) {
            v16bf bv = __builtin_bit_cast(v16bf, *(const v16u*)&sB[cur][t][lane][0]);
            acc[t] = __builtin_amdgcn_wmma_f32_16x16x32_bf16(
                false, av, false, bv, (short)0, acc[t], false, false);
        }
    }
    const int OH = 2 * IH, OW = 2 * IW;
#pragma unroll
    for (int r = 0; r < 8; ++r) {
        int m = mb * 128 + wave * 16 + hi8 + r;
        if (m >= Cout) continue;
        float bvl = bias ? bias[m] : 0.f;
#pragma unroll
        for (int t = 0; t < 4; ++t) {
            int p = nbase + t * 16 + lm;
            if (p >= Ntc) continue;
            int b = p / IHW; int r2 = p - b * IHW;
            int i2 = r2 / IW; int j = r2 - i2 * IW;
            int oh = 2 * i2 + py, ow = 2 * j + px;
            size_t off = (((size_t)b * Cout + m) * OH + oh) * OW + ow;
            float v = acc[t][r] + bvl;
            if (postact == 1) v = lrelu(v);
            else if (postact == 3) v = tanhf(v);
            Y[off] = v;
        }
    }
}

// ---------------------------------------------------------------------------
// VQ: codebook row norms
// ---------------------------------------------------------------------------
__global__ void vq_cnorm(const float* __restrict__ CB, float* __restrict__ cn,
                         int Kcb, int D)
{
    int j = blockIdx.x * blockDim.x + threadIdx.x;
    if (j >= Kcb) return;
    const float* row = CB + (size_t)j * D;
    float s = 0.f;
    for (int d = 0; d < D; ++d) { float v = row[d]; s += v * v; }
    cn[j] = s;
}

// ---------------------------------------------------------------------------
// VQ argmin: dist = ||c||^2 - 2 z.c. Packed codebook (64KB bf16, fragment
// order) is staged into LDS once per block by the Tensor Data Mover; the hot
// loop is then 2 LDS fragment reads + 2 WMMAs per 16-code tile. Argmin
// reduction through LDS with explicit s_wait_dscnt.
// ---------------------------------------------------------------------------
__global__ __launch_bounds__(256) void vq_argmin_wmma(
    const float* __restrict__ Z, const unsigned short* __restrict__ CBp,
    const float* __restrict__ cn, int* __restrict__ inds,
    int Npts, int Kcb, int HW, int DHW)
{
    __shared__ __align__(32) unsigned short sCB[32768];   // 64KB packed codebook
    __shared__ float sd[8][16][16];
    const int lane = threadIdx.x & 31;
    const int wave = threadIdx.x >> 5;
    const int gw = blockIdx.x * 8 + wave;
    const int laneHi = lane >> 4, lm = lane & 15, hi8 = laneHi * 8;
    const int nElem = (Kcb / 16) * 2 * 512;               // bf16 elements

    // ---- TDM: wave 0 issues one 1-D tensor load (8192 x 8B) global -> LDS ----
    if (wave == 0) {
        unsigned int ldsOff = (unsigned int)(unsigned long long)(void*)&sCB[0];
        unsigned long long ga = (unsigned long long)CBp;
        unsigned int n8 = (unsigned int)(nElem / 4);      // 8-byte elements
        // D# group0: count=1 | lds_addr | global_addr[56:0] | type=2  (ISA 8.3)
        u32x4 g0 = { 1u, ldsOff,
                     (unsigned int)(ga & 0xFFFFFFFFu),
                     (unsigned int)((ga >> 32) & 0x01FFFFFFu) | (2u << 30) };
        // D# group1: data_size=8B; tensor_dim0 = tile_dim0 = n8; stride = n8 (ISA 8.4)
        i32x8 g1 = { (int)(3u << 16),              // data_size = 3 (8B)
                     (int)((n8 & 0xFFFFu) << 16),  // tensor_dim0[15:0] @ bits63:48
                     (int)(n8 >> 16),              // tensor_dim0[31:16] @ bits79:64
                     (int)((n8 & 0xFFFFu) << 16),  // tile_dim0 @ bits127:112
                     0,                            // tile_dim1/2 unused (1-D)
                     (int)n8,                      // tensor_dim0_stride[31:0]
                     0, 0 };
        i32x4 g2 = { 0, 0, 0, 0 };
        i32x4 g3 = { 0, 0, 0, 0 };
#if defined(__clang_major__) && (__clang_major__ >= 23)
        i32x8 g4 = { 0, 0, 0, 0, 0, 0, 0, 0 };
        __builtin_amdgcn_tensor_load_to_lds(g0, g1, g2, g3, g4, 0);
#else
        __builtin_amdgcn_tensor_load_to_lds(g0, g1, g2, g3, 0);
#endif
        __builtin_amdgcn_s_wait_tensorcnt(0);
    }

    // ---- A fragments (z points) overlapped with the TDM transfer ----
    int p = gw * 16 + lm;
    int b = p / HW; int hw = p - b * HW;
    const float* zrow = Z + (size_t)b * DHW + hw;   // stride HW over dims
    v16u a0u, a1u;
#pragma unroll
    for (int h = 0; h < 16; ++h) {
        int d0 = a_k(h, hi8);
        a0u[h] = f2bf(zrow[(size_t)d0 * HW]);
        a1u[h] = f2bf(zrow[(size_t)(d0 + 32) * HW]);
    }
    v16bf a0 = __builtin_bit_cast(v16bf, a0u);
    v16bf a1 = __builtin_bit_cast(v16bf, a1u);

    __syncthreads();                                 // codebook resident in LDS
    if (gw * 16 >= Npts) return;

    const v8f vzero = {0.f,0.f,0.f,0.f,0.f,0.f,0.f,0.f};
    float bestd = 3.4e38f; int besti = 0;

    for (int ct = 0; ct < Kcb / 16; ++ct) {
        size_t cb0 = ((size_t)ct * 2) * 512 + (size_t)lane * 16;
        v16bf b0 = __builtin_bit_cast(v16bf, *(const v16u*)&sCB[cb0]);
        v16bf b1 = __builtin_bit_cast(v16bf, *(const v16u*)&sCB[cb0 + 512]);
        v8f acc = vzero;
        acc = __builtin_amdgcn_wmma_f32_16x16x32_bf16(false, a0, false, b0, (short)0, acc, false, false);
        acc = __builtin_amdgcn_wmma_f32_16x16x32_bf16(false, a1, false, b1, (short)0, acc, false, false);
        float cnv = cn[ct * 16 + lm];
#pragma unroll
        for (int r = 0; r < 8; ++r)
            sd[wave][r + hi8][lm] = cnv - 2.0f * acc[r];
        asm volatile("s_wait_dscnt 0" ::: "memory");
        if (lane < 16) {
#pragma unroll
            for (int n = 0; n < 16; ++n) {
                float dv = sd[wave][lane][n];
                if (dv < bestd) { bestd = dv; besti = ct * 16 + n; }
            }
        }
    }
    if (lane < 16 && gw * 16 + lane < Npts)
        inds[gw * 16 + lane] = besti;
}

// ---------------------------------------------------------------------------
// Gather quantized latents + accumulate sum((q - z)^2)
// ---------------------------------------------------------------------------
__global__ void vq_gather_loss(const float* __restrict__ Z, const float* __restrict__ CB,
                               const int* __restrict__ inds, float* __restrict__ Q,
                               float* __restrict__ lossacc, int total, int D, int HW)
{
    __shared__ float red[256];
    int t = blockIdx.x * 256 + threadIdx.x;
    float sq = 0.f;
    if (t < total) {
        int b = t / (D * HW);
        int r = t - b * (D * HW);
        int d = r / HW;
        int hw = r - d * HW;
        int p = b * HW + hw;
        float qv = CB[(size_t)inds[p] * D + d];
        float diff = qv - Z[t];
        Q[t] = qv;
        sq = diff * diff;
    }
    red[threadIdx.x] = sq;
    __syncthreads();
    for (int s = 128; s > 0; s >>= 1) {
        if (threadIdx.x < s) red[threadIdx.x] += red[threadIdx.x + s];
        __syncthreads();
    }
    if (threadIdx.x == 0) atomicAdd(lossacc, red[0]);
}

__global__ void zero1(float* p) { p[0] = 0.f; }

__global__ void vq_finalize(const float* __restrict__ lossacc, float* __restrict__ out,
                            int off, float inv)
{
    float L = lossacc[0] * inv;
    out[off] = L;       // embedding_loss
    out[off + 1] = L;   // commitment_loss (equal in forward value)
}

// ---------------------------------------------------------------------------
// Host orchestration
// ---------------------------------------------------------------------------
static inline unsigned short* pack_conv(hipStream_t st, const float* W, unsigned short* P,
                                        int Cout, int KK)
{
    int mtPad = ((Cout + 127) / 128) * 8;
    int kSteps = (KK + 31) / 32;
    size_t total = (size_t)mtPad * kSteps * 512;
    pack_conv_w<<<(unsigned)((total + 255) / 256), 256, 0, st>>>(W, P, Cout, KK, mtPad, kSteps);
    return P + total;
}

static inline unsigned short* pack_tconv(hipStream_t st, const float* W, unsigned short* P,
                                         int Cin, int Cout)
{
    int mtPad = ((Cout + 127) / 128) * 8;
    int kSteps = (Cin * 4 + 31) / 32;
    size_t total = (size_t)4 * mtPad * kSteps * 512;
    pack_tconv_w<<<(unsigned)((total + 255) / 256), 256, 0, st>>>(W, P, Cin, Cout, mtPad, kSteps);
    return P + total;
}

static inline void run_conv(hipStream_t st, const float* X, const unsigned short* Wp,
                            const float* bias, const float* res, float* Y,
                            int B, int Cin, int IH, int IW, int Cout, int OH, int OW,
                            int KH, int KW, int stride, int pad, int preact, int postact)
{
    int Ntot = B * OH * OW;
    int blocks = ((Cout + 127) / 128) * (Ntot / 64);
    conv_wmma<<<blocks, 256, 0, st>>>(X, Wp, bias, res, Y, B, Cin, IH, IW, Cout, OH, OW,
                                      KH, KW, stride, pad, preact, postact);
}

static inline void run_tconv(hipStream_t st, const float* X, const unsigned short* Wp,
                             const float* bias, float* Y,
                             int B, int Cin, int IH, int IW, int Cout,
                             int preact, int postact)
{
    int Ntc = B * IH * IW;
    int mtPad = ((Cout + 127) / 128) * 8;
    dim3 grid(((Cout + 127) / 128) * (Ntc / 64), 4);
    tconv_wmma<<<grid, 256, 0, st>>>(X, Wp, bias, Y, B, Cin, IH, IW, Cout, mtPad,
                                     preact, postact);
}

extern "C" void kernel_launch(void* const* d_in, const int* in_sizes, int n_in,
                              void* d_out, int out_size, void* d_ws, size_t ws_size,
                              hipStream_t stream)
{
    (void)in_sizes; (void)n_in; (void)ws_size;
    const float* x    = (const float*)d_in[0];
    const float* ew0  = (const float*)d_in[1];
    const float* eb0  = (const float*)d_in[2];
    const float* ew1  = (const float*)d_in[3];
    const float* eb1  = (const float*)d_in[4];
    const float* ew2  = (const float*)d_in[5];
    const float* eb2  = (const float*)d_in[6];
    const float* erw3 = (const float*)d_in[7];
    const float* erw1 = (const float*)d_in[8];
    const float* ew3  = (const float*)d_in[9];
    const float* eb3  = (const float*)d_in[10];
    const float* cb   = (const float*)d_in[11];
    const float* dw0  = (const float*)d_in[12];
    const float* db0  = (const float*)d_in[13];
    const float* drw3 = (const float*)d_in[14];
    const float* drw1 = (const float*)d_in[15];
    const float* tw0  = (const float*)d_in[16];
    const float* tb0  = (const float*)d_in[17];
    const float* tw1  = (const float*)d_in[18];
    const float* tb1  = (const float*)d_in[19];
    float* out = (float*)d_out;

    const int Bsz = 16, H0 = 128, H1 = 256, D = 64, K = 512, NRES = 6;
    const int HW32 = 32 * 32;
    const int Npts = Bsz * HW32;           // 16384

    // ---- workspace: fp32 activations, then 32B-aligned packed-weight region
    float* A0 = (float*)d_ws;              // [16,128,64,64] = 8388608
    float* B1 = A0 + 8388608;              // [16,256,32,32] = 4194304
    float* B2 = B1 + 4194304;
    float* B3 = B2 + 4194304;
    float* Zb = B3 + 4194304;              // [16, 64,32,32] = 1048576
    float* Qb = Zb + 1048576;              // [16, 64,32,32]
    int*   inds = (int*)(Qb + 1048576);    // 16384
    float* cn = (float*)(inds + 16384);    // 512
    float* lossacc = cn + 512;             // 1
    uintptr_t up = (uintptr_t)(lossacc + 1);
    up = (up + 31) & ~(uintptr_t)31;
    unsigned short* wc = (unsigned short*)up;

    // ---- pack all weights (bf16, fragment-ordered); in-order on stream ----
    unsigned short* p_ew0 = wc;  wc = pack_conv(stream, ew0, wc, H0, 3 * 16);
    unsigned short* p_ew1 = wc;  wc = pack_conv(stream, ew1, wc, H1, H0 * 16);
    unsigned short* p_ew2 = wc;  wc = pack_conv(stream, ew2, wc, H1, H1 * 9);
    unsigned short* p_er3[NRES], *p_er1[NRES], *p_dr3[NRES], *p_dr1[NRES];
    for (int i = 0; i < NRES; ++i) {
        p_er3[i] = wc; wc = pack_conv(stream, erw3 + (size_t)i * H1 * H1 * 9, wc, H1, H1 * 9);
        p_er1[i] = wc; wc = pack_conv(stream, erw1 + (size_t)i * H1 * H1,     wc, H1, H1);
    }
    unsigned short* p_ew3 = wc;  wc = pack_conv(stream, ew3, wc, D, H1);
    unsigned short* p_dw0 = wc;  wc = pack_conv(stream, dw0, wc, H1, D * 9);
    for (int i = 0; i < NRES; ++i) {
        p_dr3[i] = wc; wc = pack_conv(stream, drw3 + (size_t)i * H1 * H1 * 9, wc, H1, H1 * 9);
        p_dr1[i] = wc; wc = pack_conv(stream, drw1 + (size_t)i * H1 * H1,     wc, H1, H1);
    }
    unsigned short* p_tw0 = wc;  wc = pack_tconv(stream, tw0, wc, H1, H0);
    unsigned short* p_tw1 = wc;  wc = pack_tconv(stream, tw1, wc, H0, 3);
    unsigned short* p_cb  = wc;
    pack_cb<<<((K / 16) * 2 * 512 + 255) / 256, 256, 0, stream>>>(cb, p_cb, K, D);

    // ---------------- encoder ----------------
    run_conv(stream, x,  p_ew0, eb0, nullptr, A0, Bsz,   3, 128, 128, H0, 64, 64, 4, 4, 2, 1, 0, 1);
    run_conv(stream, A0, p_ew1, eb1, nullptr, B1, Bsz,  H0,  64,  64, H1, 32, 32, 4, 4, 2, 1, 0, 1);
    run_conv(stream, B1, p_ew2, eb2, nullptr, B2, Bsz,  H1,  32,  32, H1, 32, 32, 3, 3, 1, 1, 0, 1);
    {   // residual stack: x += conv1x1(relu(conv3x3(x)))
        float* Xb = B2; float* Yb = B1; float* Hb = B3;
        for (int i = 0; i < NRES; ++i) {
            run_conv(stream, Xb, p_er3[i], nullptr, nullptr, Hb, Bsz, H1, 32, 32, H1, 32, 32, 3, 3, 1, 1, 0, 2);
            run_conv(stream, Hb, p_er1[i], nullptr, Xb,      Yb, Bsz, H1, 32, 32, H1, 32, 32, 1, 1, 1, 0, 0, 0);
            float* t = Xb; Xb = Yb; Yb = t;
        }
        // NRES even -> result in B2
    }
    // z = lrelu(conv1x1(lrelu(h)))   (pre-activation fused into gather)
    run_conv(stream, B2, p_ew3, eb3, nullptr, Zb, Bsz, H1, 32, 32, D, 32, 32, 1, 1, 1, 0, 1, 1);

    // ---------------- vector quantization ----------------
    vq_cnorm<<<(K + 255) / 256, 256, 0, stream>>>(cb, cn, K, D);
    zero1<<<1, 1, 0, stream>>>(lossacc);
    vq_argmin_wmma<<<(Npts / 16 + 7) / 8, 256, 0, stream>>>(Zb, p_cb, cn, inds,
                                                            Npts, K, HW32, D * HW32);
    {
        int total = Bsz * D * HW32;        // 1048576
        vq_gather_loss<<<(total + 255) / 256, 256, 0, stream>>>(Zb, cb, inds, Qb,
                                                                lossacc, total, D, HW32);
        vq_finalize<<<1, 1, 0, stream>>>(lossacc, out, out_size - 2, 1.0f / (float)total);
    }

    // ---------------- decoder ----------------
    run_conv(stream, Qb, p_dw0, db0, nullptr, B1, Bsz, D, 32, 32, H1, 32, 32, 3, 3, 1, 1, 0, 1);
    {
        float* Xb = B1; float* Yb = B2; float* Hb = B3;
        for (int i = 0; i < NRES; ++i) {
            run_conv(stream, Xb, p_dr3[i], nullptr, nullptr, Hb, Bsz, H1, 32, 32, H1, 32, 32, 3, 3, 1, 1, 0, 2);
            run_conv(stream, Hb, p_dr1[i], nullptr, Xb,      Yb, Bsz, H1, 32, 32, H1, 32, 32, 1, 1, 1, 0, 0, 0);
            float* t = Xb; Xb = Yb; Yb = t;
        }
        // result in B1
    }
    // lrelu(tconv) with fused lrelu on input: -> [16,128,64,64]
    run_tconv(stream, B1, p_tw0, tb0, A0, Bsz, H1, 32, 32, H0, 1, 1);
    // tanh(tconv): -> [16,3,128,128] written directly to d_out
    run_tconv(stream, A0, p_tw1, tb1, out, Bsz, H0, 64, 64, 3, 0, 3);
}
